// Attention_403726926318
// MI455X (gfx1250) — matrix operand order repaired and checked by reading, not compiled, not run on hardware
//
#include <hip/hip_runtime.h>
#include <hip/hip_bf16.h>

// ---------------- CDNA5 WMMA plumbing (wave32) ----------------
typedef __attribute__((ext_vector_type(16))) __bf16       v16bf;
typedef __attribute__((ext_vector_type(8)))  float        v8f;
typedef __attribute__((ext_vector_type(4)))  unsigned int u32x4;

union Frag {            // one 16x32 (A) or 32x16 (B) bf16 operand: 8 VGPRs/lane
    v16bf v;
    u32x4 q[2];
};

__device__ __forceinline__ v8f wmma_bf16(v16bf a, v16bf b, v8f c) {
    // D(f32 16x16) = A(bf16 16x32) * B(bf16 32x16) + C
    return __builtin_amdgcn_wmma_f32_16x16x32_bf16(
        /*neg_a=*/false, a, /*neg_b=*/false, b,
        /*c_mod=*/(short)0, c, /*reuse_a=*/false, /*reuse_b=*/false);
}

// Direct global->LDS async copy (16B per lane), tracked on ASYNCcnt.
// Per ISA 10.2 aperture rules, the low 32 bits of a generic LDS address are the
// LDS byte offset, so truncating the generic pointer gives the VDST operand.
__device__ __forceinline__ void async_b128(void* dst_lds, const void* src_global) {
    unsigned loff = (unsigned)(unsigned long long)dst_lds;
    asm volatile("global_load_async_to_lds_b128 %0, %1, off"
                 :: "v"(loff), "v"(src_global) : "memory");
}
__device__ __forceinline__ void wait_async() {
    asm volatile("s_wait_asynccnt 0x0" ::: "memory");
}

#define HEADS 16
#define DHEAD 64
#define SEQ   2048
#define BATCH 2
#define ROWS  (BATCH * SEQ)   // 4096
#define KDIM  1024
#define NQKV  3072
#define SCALE 0.125f          // 64^-0.5

// ---------------- fp32 -> bf16 conversions ----------------
__global__ void cvt_x_kernel(const float* __restrict__ x, __bf16* __restrict__ xb, int n) {
    int i = blockIdx.x * blockDim.x + threadIdx.x;
    int stride = gridDim.x * blockDim.x;
    for (; i < n; i += stride) xb[i] = (__bf16)x[i];
}

// w: [K][N] fp32 row-major  ->  wT: [N][K] bf16 (so GEMM B-tiles are K-contiguous)
__global__ void cvt_transpose_kernel(const float* __restrict__ w, __bf16* __restrict__ wT,
                                     int K, int N) {
    int i = blockIdx.x * blockDim.x + threadIdx.x;
    int total = K * N;
    int stride = gridDim.x * blockDim.x;
    for (; i < total; i += stride) {
        int n = i / K, k = i - n * K;
        wT[i] = (__bf16)w[(size_t)k * N + n];
    }
}

// ---------------- tiled bf16 GEMM: C[128x64 tile] = A[.][1024] x B^T ----------------
// Double-buffered async-to-LDS pipeline: tile i+1 DMA in flight during tile i WMMAs.
// mode 0: QKV epilogue (scale Q, mask K, transpose V)   mode 1: plain fp32 store
__launch_bounds__(256)
__global__ void gemm_bf16_kernel(const __bf16* __restrict__ A,   // [ROWS][1024]
                                 const __bf16* __restrict__ B,   // [Ntot][1024]
                                 int mode,
                                 const float*  __restrict__ mask, // [BATCH*SEQ]
                                 __bf16* __restrict__ Qw,         // [32][2048][64]
                                 __bf16* __restrict__ Kw,         // [32][2048][64]
                                 __bf16* __restrict__ Vtw,        // [32][64][2048]
                                 float*  __restrict__ Cf)         // [ROWS][1024]
{
    __shared__ __bf16 At[2][128][64];   // 32 KB (ping-pong)
    __shared__ __bf16 Bt[2][64][64];    // 16 KB (ping-pong)

    const int tid  = threadIdx.x;
    const int wave = tid >> 5;
    const int lane = tid & 31;
    const int hf   = lane >> 4;
    const int l16  = lane & 15;
    const int m0   = blockIdx.x * 128;
    const int n0   = blockIdx.y * 64;

    auto stage = [&](int buf, int k0) {
        #pragma unroll
        for (int it = 0; it < 4; ++it) {          // A tile 128x64: 1024 x 16B
            int idx = tid + it * 256;
            int r = idx >> 3, c = idx & 7;
            async_b128(&At[buf][r][c * 8], &A[(size_t)(m0 + r) * KDIM + k0 + c * 8]);
        }
        #pragma unroll
        for (int it = 0; it < 2; ++it) {          // B tile 64x64: 512 x 16B
            int idx = tid + it * 256;
            int r = idx >> 3, c = idx & 7;
            async_b128(&Bt[buf][r][c * 8], &B[(size_t)(n0 + r) * KDIM + k0 + c * 8]);
        }
    };

    v8f zero8 = {0.f, 0.f, 0.f, 0.f, 0.f, 0.f, 0.f, 0.f};
    v8f acc[4] = {zero8, zero8, zero8, zero8};

    stage(0, 0);
    wait_async();
    __syncthreads();

    const int ar = wave * 16 + l16;
    for (int k0 = 0; k0 < KDIM; k0 += 64) {
        const int cur = (k0 >> 6) & 1;
        if (k0 + 64 < KDIM) stage(cur ^ 1, k0 + 64);   // prefetch next tile (async)

        #pragma unroll
        for (int kd = 0; kd < 2; ++kd) {
            Frag a;
            a.q[0] = *(const u32x4*)&At[cur][ar][kd * 32 + hf * 8];
            a.q[1] = *(const u32x4*)&At[cur][ar][kd * 32 + 16 + hf * 8];
            Frag b[4];                 // distinct regs -> partial DS waits, pipelined WMMA
            #pragma unroll
            for (int ng = 0; ng < 4; ++ng) {
                int br = ng * 16 + l16;
                b[ng].q[0] = *(const u32x4*)&Bt[cur][br][kd * 32 + hf * 8];
                b[ng].q[1] = *(const u32x4*)&Bt[cur][br][kd * 32 + 16 + hf * 8];
            }
            #pragma unroll
            for (int ng = 0; ng < 4; ++ng)
                acc[ng] = wmma_bf16(a.v, b[ng].v, acc[ng]);
        }

        wait_async();      // next tile's DMA done (own wave)
        __syncthreads();   // all waves done reading cur / writing nxt
    }

    // epilogue: C rows m = wave*16 + r + 8*hf, col = n0 + ng*16 + l16
    #pragma unroll
    for (int ng = 0; ng < 4; ++ng) {
        int n = n0 + ng * 16 + l16;
        #pragma unroll
        for (int r = 0; r < 8; ++r) {
            int gm = m0 + wave * 16 + r + hf * 8;
            float v = acc[ng][r];
            if (mode == 0) {
                int bb  = gm >> 11;          // / 2048
                int seq = gm & 2047;
                int which = n >> 10;         // q / k / v
                int hh  = (n & 1023) >> 6;
                int d   = n & 63;
                size_t bh = (size_t)(bb * HEADS + hh);
                if (which == 0) {
                    Qw[(bh * SEQ + seq) * DHEAD + d] = (__bf16)(v * SCALE);
                } else if (which == 1) {
                    // fold multiplicative key mask into K rows
                    Kw[(bh * SEQ + seq) * DHEAD + d] = (__bf16)(v * mask[bb * SEQ + seq]);
                } else {
                    Vtw[(bh * DHEAD + d) * SEQ + seq] = (__bf16)v;  // V transposed
                }
            } else {
                Cf[(size_t)gm * KDIM + n] = v;
            }
        }
    }
}

// ---------------- fused flash attention ----------------
// grid: (SEQ/128 q-tiles, BATCH*HEADS). 8 waves, each owns 16 query rows.
// Double-buffered K/V tiles streamed via async-to-LDS.
__launch_bounds__(256)
__global__ void flash_attn_kernel(const __bf16* __restrict__ Qw,
                                  const __bf16* __restrict__ Kw,
                                  const __bf16* __restrict__ Vtw,
                                  __bf16* __restrict__ Ow)   // [b][seq][h*64+d]
{
    __shared__ __bf16 Kt[2][32][64];    // 8 KB (ping-pong)
    __shared__ __bf16 Vt[2][64][32];    // 8 KB (ping-pong)
    __shared__ __bf16 Pb[8][16][32];    // 8 KB per-wave P bounce buffer

    const int tid  = threadIdx.x;
    const int wave = tid >> 5;
    const int lane = tid & 31;
    const int hf   = lane >> 4;
    const int l16  = lane & 15;
    const int bh   = blockIdx.y;
    const int q0   = blockIdx.x * 128;
    const int b    = bh >> 4;
    const int h    = bh & 15;

    auto stage = [&](int buf, int j0) {
        {   // K tile 32x64 (4 KB)
            int r = tid >> 3, c = tid & 7;
            async_b128(&Kt[buf][r][c * 8], &Kw[((size_t)bh * SEQ + j0 + r) * DHEAD + c * 8]);
        }
        {   // V^T tile 64x32 (4 KB)
            int r = tid >> 2, c = tid & 3;
            async_b128(&Vt[buf][r][c * 8], &Vtw[((size_t)bh * DHEAD + r) * SEQ + j0 + c * 8]);
        }
    };

    // Q fragments live in registers for the whole key loop (scale pre-folded)
    Frag qf[2];
    {
        const __bf16* qrow = Qw + ((size_t)bh * SEQ + q0 + wave * 16 + l16) * DHEAD;
        #pragma unroll
        for (int kd = 0; kd < 2; ++kd) {
            qf[kd].q[0] = *(const u32x4*)(qrow + kd * 32 + hf * 8);
            qf[kd].q[1] = *(const u32x4*)(qrow + kd * 32 + 16 + hf * 8);
        }
    }

    v8f zero8 = {0.f, 0.f, 0.f, 0.f, 0.f, 0.f, 0.f, 0.f};
    v8f o[4] = {zero8, zero8, zero8, zero8};
    float rmax[8], rsum[8];
    #pragma unroll
    for (int r = 0; r < 8; ++r) { rmax[r] = -1e30f; rsum[r] = 0.f; }

    stage(0, 0);
    wait_async();
    __syncthreads();

    for (int j0 = 0; j0 < SEQ; j0 += 32) {
        const int cur = (j0 >> 5) & 1;
        if (j0 + 32 < SEQ) stage(cur ^ 1, j0 + 32);   // prefetch next K/V tile (async)

        // S = Q K^T for two 16-key column groups
        v8f s[2];
        #pragma unroll
        for (int cg = 0; cg < 2; ++cg) {
            v8f sc = zero8;
            #pragma unroll
            for (int kd = 0; kd < 2; ++kd) {
                Frag bfr;
                int br = cg * 16 + l16;
                bfr.q[0] = *(const u32x4*)&Kt[cur][br][kd * 32 + hf * 8];
                bfr.q[1] = *(const u32x4*)&Kt[cur][br][kd * 32 + 16 + hf * 8];
                sc = wmma_bf16(qf[kd].v, bfr.v, sc);
            }
            s[cg] = sc;
        }

        // online softmax (fp32) over the 32 keys; row = r + 8*hf, col = lane%16 (+16)
        float alpha[8];
        #pragma unroll
        for (int r = 0; r < 8; ++r) {
            float mloc = fmaxf(s[0][r], s[1][r]);
            #pragma unroll
            for (int off = 8; off >= 1; off >>= 1)
                mloc = fmaxf(mloc, __shfl_xor(mloc, off, 16));
            float mnew = fmaxf(rmax[r], mloc);
            float p0 = __expf(s[0][r] - mnew);
            float p1 = __expf(s[1][r] - mnew);
            float psum = p0 + p1;
            #pragma unroll
            for (int off = 8; off >= 1; off >>= 1)
                psum += __shfl_xor(psum, off, 16);
            alpha[r] = __expf(rmax[r] - mnew);
            rsum[r]  = rsum[r] * alpha[r] + psum;
            rmax[r]  = mnew;
            // C-layout -> A-layout via wave-private LDS
            Pb[wave][r + hf * 8][l16]      = (__bf16)p0;
            Pb[wave][r + hf * 8][16 + l16] = (__bf16)p1;
        }
        // cross-lane LDS dependency within the wave: drain DS ops
        asm volatile("s_wait_dscnt 0x0" ::: "memory");

        #pragma unroll
        for (int dg = 0; dg < 4; ++dg)
            #pragma unroll
            for (int r = 0; r < 8; ++r)
                o[dg][r] *= alpha[r];

        Frag pa;   // A-fragment of P (16x32)
        pa.q[0] = *(const u32x4*)&Pb[wave][l16][hf * 8];
        pa.q[1] = *(const u32x4*)&Pb[wave][l16][16 + hf * 8];
        Frag vb[4];
        #pragma unroll
        for (int dg = 0; dg < 4; ++dg) {
            int br = dg * 16 + l16;
            vb[dg].q[0] = *(const u32x4*)&Vt[cur][br][hf * 8];
            vb[dg].q[1] = *(const u32x4*)&Vt[cur][br][16 + hf * 8];
        }
        #pragma unroll
        for (int dg = 0; dg < 4; ++dg)
            o[dg] = wmma_bf16(pa.v, vb[dg].v, o[dg]);

        wait_async();      // next tile's DMA done (own wave)
        __syncthreads();   // all waves done reading cur / writing nxt
    }

    // normalize and store O as [b][seq][h*64+d] (ready for out-proj GEMM)
    #pragma unroll
    for (int dg = 0; dg < 4; ++dg) {
        #pragma unroll
        for (int r = 0; r < 8; ++r) {
            int seq = q0 + wave * 16 + r + hf * 8;
            float v = o[dg][r] / rsum[r];
            Ow[((size_t)b * SEQ + seq) * KDIM + h * 64 + dg * 16 + l16] = (__bf16)v;
        }
    }
}

// ---------------- host launcher ----------------
extern "C" void kernel_launch(void* const* d_in, const int* in_sizes, int n_in,
                              void* d_out, int out_size, void* d_ws, size_t ws_size,
                              hipStream_t stream) {
    const float* x     = (const float*)d_in[0];  // [2][2048][1024]
    const float* smask = (const float*)d_in[1];  // [2][2048]
    const float* w_qkv = (const float*)d_in[2];  // [1024][3072]
    const float* w_out = (const float*)d_in[3];  // [1024][1024]
    float* out = (float*)d_out;                  // [2][2048][1024]

    char* ws = (char*)d_ws;
    size_t off = 0;
    auto alloc = [&](size_t bytes) {
        void* p = ws + off;
        off += (bytes + 255) & ~(size_t)255;
        return p;
    };
    __bf16* xb  = (__bf16*)alloc((size_t)ROWS * KDIM * 2);      // 8 MB
    __bf16* wqT = (__bf16*)alloc((size_t)NQKV * KDIM * 2);      // 6 MB
    __bf16* woT = (__bf16*)alloc((size_t)KDIM * KDIM * 2);      // 2 MB
    __bf16* Qw  = (__bf16*)alloc((size_t)BATCH * HEADS * SEQ * DHEAD * 2);  // 8 MB
    __bf16* Kw  = (__bf16*)alloc((size_t)BATCH * HEADS * SEQ * DHEAD * 2);  // 8 MB
    __bf16* Vtw = (__bf16*)alloc((size_t)BATCH * HEADS * DHEAD * SEQ * 2);  // 8 MB
    __bf16* Ow  = (__bf16*)alloc((size_t)ROWS * KDIM * 2);      // 8 MB
    (void)ws_size; (void)in_sizes; (void)n_in; (void)out_size;

    cvt_x_kernel<<<2048, 256, 0, stream>>>(x, xb, ROWS * KDIM);
    cvt_transpose_kernel<<<4096, 256, 0, stream>>>(w_qkv, wqT, KDIM, NQKV);
    cvt_transpose_kernel<<<2048, 256, 0, stream>>>(w_out, woT, KDIM, KDIM);

    // QKV projection: [4096 x 1024] x [1024 x 3072], fused scale/mask/transpose epilogue
    gemm_bf16_kernel<<<dim3(ROWS / 128, NQKV / 64), 256, 0, stream>>>(
        xb, wqT, 0, smask, Qw, Kw, Vtw, nullptr);

    // fused attention: 16 q-tiles x 32 (b,h)
    flash_attn_kernel<<<dim3(SEQ / 128, BATCH * HEADS), 256, 0, stream>>>(Qw, Kw, Vtw, Ow);

    // output projection: [4096 x 1024] x [1024 x 1024] -> fp32 out
    gemm_bf16_kernel<<<dim3(ROWS / 128, KDIM / 64), 256, 0, stream>>>(
        Ow, woT, 1, nullptr, nullptr, nullptr, nullptr, out);
}
